// MotionGenerator_RNN_3831110828510
// MI455X (gfx1250) — matrix operand order repaired and checked by reading, not compiled, not run on hardware
//
#include <hip/hip_runtime.h>

// ---------------------------------------------------------------------------
// MI455X (gfx1250) implementation.
//  * bf16 weights (pre-transposed [N][Kpad]) + v_wmma_f32_16x16x32_bf16,
//    fp32 accumulate. 28MB of bf16 weights stay resident in the 192MB L2,
//    so per-step weight traffic never touches HBM after step 0.
//  * Persistent scan kernel: 32 blocks x 256 threads (8 wave32 each),
//    5 device-scope barriers per timestep, GRU gate columns owned per-block
//    so the nonlinearity needs no extra sync.
//  * xcat staged in LDS (48KB of the 320KB/WGP) and consumed directly as
//    WMMA A-fragments.
// ---------------------------------------------------------------------------

typedef __attribute__((ext_vector_type(16))) __bf16 v16bf;
typedef __attribute__((ext_vector_type(8)))  __bf16 v8bf;
typedef __attribute__((ext_vector_type(8)))  float  v8f;

constexpr int kT    = 448;   // timesteps
constexpr int kXPAD = 384;   // xcat 382 -> padded
constexpr int kGPAD = 1408;  // gx 1406 -> padded
constexpr int NB    = 32;    // persistent blocks
constexpr int TPB   = 256;   // 8 wave32

__device__ inline float gelu_f(float x) {
  return 0.5f * x * (1.0f + erff(x * 0.70710678118654752f));
}
__device__ inline float sigm_f(float x) { return 1.0f / (1.0f + expf(-x)); }

// A fragment: 16x32 bf16 tile at (r0,k0) of row-major A (leading dim lda).
// ISA layout: lanes 0-15 -> rows 0-15 K{0..7,16..23}; lanes 16-31 -> K{8..15,24..31}.
__device__ inline v16bf load_a16(const __bf16* A, int lda, int r0, int k0, int lane) {
  int r = lane & 15, hi = lane >> 4;
  const __bf16* p = A + (size_t)(r0 + r) * lda + k0 + hi * 8;
  v8bf x = *(const v8bf*)p;
  v8bf y = *(const v8bf*)(p + 16);
  v16bf o;
#pragma unroll
  for (int i = 0; i < 8; ++i) { o[i] = x[i]; o[i + 8] = y[i]; }
  return o;
}

// B fragment: 32x16 tile; weights stored transposed WT[n][k] (leading dim ldk).
// lanes 0-15 -> col n0+lane, K{0..7,16..23} ; lanes 16-31 -> col n0+lane-16, K{8..15,24..31}.
__device__ inline v16bf load_b16(const __bf16* WT, int ldk, int n0, int k0, int lane) {
  int c = lane & 15, hk = (lane >> 4) << 3;
  const __bf16* p = WT + (size_t)(n0 + c) * ldk + k0 + hk;
  v8bf x = *(const v8bf*)p;
  v8bf y = *(const v8bf*)(p + 16);
  v16bf o;
#pragma unroll
  for (int i = 0; i < 8; ++i) { o[i] = x[i]; o[i + 8] = y[i]; }
  return o;
}

__device__ inline v8f wmma_bf16(v16bf a, v16bf b, v8f c) {
  return __builtin_amdgcn_wmma_f32_16x16x32_bf16(false, a, false, b, (short)0, c,
                                                 false, false);
}

// Device-scope two-phase barrier across NB persistent blocks.
__device__ inline void grid_sync(int* bar) {
  __syncthreads();
  if (threadIdx.x == 0) {
    __threadfence();
    int gen = atomicAdd(&bar[1], 0);
    if (atomicAdd(&bar[0], 1) == NB - 1) {
      bar[0] = 0;
      __threadfence();
      atomicAdd(&bar[1], 1);
    } else {
      while (atomicAdd(&bar[1], 0) == gen) __builtin_amdgcn_s_sleep(1);
    }
    __threadfence();
  }
  __syncthreads();
}

// fp32 W[K][Nsrc] row-major  ->  bf16 WT[Npad][Kpad], zero padded.
__device__ inline void conv_wT(const float* W, __bf16* WT, int K, int Kpad,
                               int Nsrc, int Npad, size_t g, size_t gs) {
  size_t total = (size_t)Npad * Kpad;
  for (size_t i = g; i < total; i += gs) {
    int n = (int)(i / Kpad), k = (int)(i % Kpad);
    float v = (k < K && n < Nsrc) ? W[(size_t)k * Nsrc + n] : 0.f;
    WT[i] = (__bf16)v;
  }
}

// ---------------------------------------------------------------------------
// Kernel 1: weight conversion, x0/pre_mo init, style path (+ z_mu/z_lv outs).
// ---------------------------------------------------------------------------
__global__ __launch_bounds__(TPB) void prep_kernel(
    const float* __restrict__ mo, const float* __restrict__ lxm,
    const int* __restrict__ vid, const float* __restrict__ eps,
    const float* __restrict__ csi_w0, const float* __restrict__ csi_w1,
    const float* __restrict__ csi_w2, const float* __restrict__ emb_w,
    const float* __restrict__ wih0, const float* __restrict__ whh0,
    const float* __restrict__ wih1, const float* __restrict__ whh1,
    const float* __restrict__ pred_w, const float* __restrict__ spk,
    const float* __restrict__ mu_w, const float* __restrict__ mu_b,
    const float* __restrict__ lv_w, const float* __restrict__ lv_b,
    __bf16* wc0, __bf16* wc1, __bf16* wc2, __bf16* wemb,
    __bf16* wi0, __bf16* wh0, __bf16* wi1, __bf16* wh1, __bf16* wpred,
    __bf16* a_x0, float* pre_mo, float* style, float* out_mu, float* out_lv,
    int* bar) {
  size_t g  = (size_t)blockIdx.x * TPB + threadIdx.x;
  size_t gs = (size_t)gridDim.x * TPB;
  if (g == 0) { bar[0] = 0; bar[1] = 0; }

  conv_wT(csi_w0, wc0, 222, 224, 1024, 1024, g, gs);
  conv_wT(csi_w1, wc1, 1024, 1024, 1024, 1024, g, gs);
  conv_wT(csi_w2, wc2, 1024, 1024, 2048, 2048, g, gs);
  conv_wT(emb_w, wemb, 382, kXPAD, 1024, 1024, g, gs);
  conv_wT(wih0, wi0, 1406, kGPAD, 3072, 3072, g, gs);
  conv_wT(whh0, wh0, 1024, 1024, 3072, 3072, g, gs);
  conv_wT(wih1, wi1, 1024, 1024, 3072, 3072, g, gs);
  conv_wT(whh1, wh1, 1024, 1024, 3072, 3072, g, gs);
  conv_wT(pred_w, wpred, 1024, 1024, 126, 128, g, gs);

  // x0 = [mo[:,:,31], lxm[:,:,0]] bf16, K-padded to 224
  for (size_t i = g; i < (size_t)64 * 224; i += gs) {
    int n = (int)(i / 224), c = (int)(i % 224);
    float v = 0.f;
    if (c < 126)      v = mo[(size_t)n * 126 * 512 + (size_t)c * 512 + 31];
    else if (c < 222) v = lxm[(size_t)n * 96 * 16 + (size_t)(c - 126) * 16 + 0];
    a_x0[i] = (__bf16)v;
  }
  for (size_t i = g; i < (size_t)64 * 126; i += gs) {
    int n = (int)(i / 126), c = (int)(i % 126);
    pre_mo[i] = mo[(size_t)n * 126 * 512 + (size_t)c * 512 + 31];
  }

  // style[t][n][s] = z_mu + eps*exp(0.5*z_lv);  z_mu/z_lv -> outputs (64,32,448)
  for (size_t i = g; i < (size_t)kT * 64 * 32; i += gs) {
    int s = (int)(i & 31);
    int n = (int)((i >> 5) & 63);
    int t = (int)(i / (64 * 32));
    int b = t / 32 + 1;
    int v = vid[n * 16 + b];
    const float* z = spk + v * 32;
    float zm = mu_b[s], zl = lv_b[s];
#pragma unroll 4
    for (int k = 0; k < 32; ++k) {
      zm += z[k] * mu_w[k * 32 + s];
      zl += z[k] * lv_w[k * 32 + s];
    }
    style[i] = zm + eps[i] * expf(0.5f * zl);
    out_mu[((size_t)n * 32 + s) * kT + t] = zm;
    out_lv[((size_t)n * 32 + s) * kT + t] = zl;
  }
}

// ---------------------------------------------------------------------------
// Kernel 2: generic WMMA GEMM for the CSI init-MLP (M=64).
// mode 0: gelu -> bf16 outB (ld=ldo).  mode 1: split into h0/h1 f32+bf16.
// ---------------------------------------------------------------------------
__global__ __launch_bounds__(TPB) void csi_gemm(
    const __bf16* __restrict__ A, int lda, int nk,
    const __bf16* __restrict__ WT, const float* __restrict__ bias, int mode,
    __bf16* outB, int ldo, float* h0f, __bf16* h0b, float* h1f, __bf16* h1b) {
  int tid = threadIdx.x, lane = tid & 31, wave = tid >> 5;
  int mt = wave & 3, nt = wave >> 2;
  int n0 = blockIdx.x * 32 + nt * 16;
  int fr = (lane >> 4) << 3, fc = lane & 15;

  v8f acc = {};
  for (int kk = 0; kk < nk; ++kk) {
    v16bf a = load_a16(A, lda, mt * 16, kk * 32, lane);
    v16bf b = load_b16(WT, lda, n0, kk * 32, lane);
    acc = wmma_bf16(a, b, acc);
  }
#pragma unroll
  for (int i = 0; i < 8; ++i) {
    int row = mt * 16 + i + fr, col = n0 + fc;
    float v = acc[i] + bias[col];
    if (mode == 0) {
      outB[(size_t)row * ldo + col] = (__bf16)gelu_f(v);
    } else {
      int h = col >> 10, c = col & 1023;
      if (h == 0) { h0f[(size_t)row * 1024 + c] = v; h0b[(size_t)row * 1024 + c] = (__bf16)v; }
      else        { h1f[(size_t)row * 1024 + c] = v; h1b[(size_t)row * 1024 + c] = (__bf16)v; }
    }
  }
}

// ---------------------------------------------------------------------------
// Kernel 3: persistent 448-step GRU scan.
// ---------------------------------------------------------------------------
__global__ __launch_bounds__(TPB) void scan_kernel(
    const float* __restrict__ aud, const float* __restrict__ lxm,
    const float* __restrict__ emb_b, const float* __restrict__ bih0,
    const float* __restrict__ bhh0, const float* __restrict__ bih1,
    const float* __restrict__ bhh1, const float* __restrict__ ln_g,
    const float* __restrict__ ln_b, const float* __restrict__ pred_b,
    const __bf16* __restrict__ wemb, const __bf16* __restrict__ wi0,
    const __bf16* __restrict__ wh0, const __bf16* __restrict__ wi1,
    const __bf16* __restrict__ wh1, const __bf16* __restrict__ wpred,
    const float* __restrict__ style, float* pre_mo, __bf16* gxb,
    __bf16* h0b, __bf16* h1b, float* h0f, float* h1f, __bf16* outb,
    float* mo_out, int* bar) {
  __shared__ __align__(16) __bf16 s_xcat[64 * kXPAD];  // 48KB of 320KB/WGP
  __shared__ float s_sum[TPB], s_sq[TPB];

  const int tid = threadIdx.x, bid = blockIdx.x;
  const int lane = tid & 31, wave = tid >> 5;
  const int mt = wave & 3, nt = wave >> 2;
  const int fr = (lane >> 4) << 3, fc = lane & 15;

  for (int t = 0; t < kT; ++t) {
    const int rp = t & 1, wp = rp ^ 1;
    const int blkI = t / 32 + 1;

    // ---- Phase 0: build xcat (bf16) in LDS; publish this block's slice of
    //      the xcat columns of gx ([1024,1408)) to the shared gx buffer.
    for (int i = tid; i < 64 * kXPAD; i += TPB) {
      int n = i / kXPAD, c = i % kXPAD;
      float v = 0.f;
      if (c < 126)      v = pre_mo[n * 126 + c];
      else if (c < 254) v = aud[(size_t)n * 128 * kT + (size_t)(c - 126) * kT + t];
      else if (c < 350) v = lxm[(size_t)n * 96 * 16 + (size_t)(c - 254) * 16 + blkI];
      else if (c < 382) v = style[((size_t)t * 64 + n) * 32 + (c - 350)];
      s_xcat[i] = (__bf16)v;
    }
    __syncthreads();
    for (int i = tid; i < 64 * 12; i += TPB) {
      int n = i / 12, c = bid * 12 + (i % 12);
      gxb[(size_t)n * kGPAD + 1024 + c] = s_xcat[n * kXPAD + c];
    }

    // ---- Phase A: ine = gelu(xcat @ emb_in_w + b), block owns 32 columns.
    {
      int n0 = bid * 32 + nt * 16;
      v8f acc = {};
      for (int k0 = 0; k0 < kXPAD; k0 += 32) {
        v16bf a = load_a16(s_xcat, kXPAD, mt * 16, k0, lane);
        v16bf b = load_b16(wemb, kXPAD, n0, k0, lane);
        acc = wmma_bf16(a, b, acc);
      }
#pragma unroll
      for (int i = 0; i < 8; ++i) {
        int row = mt * 16 + i + fr, col = n0 + fc;
        gxb[(size_t)row * kGPAD + col] = (__bf16)gelu_f(acc[i] + emb_b[col]);
      }
    }
    grid_sync(bar);

    // ---- Phase B: GRU layer 0. Block owns hidden units [32b,32b+32);
    //      computes gate columns {j, j+1024, j+2048} locally -> no extra sync.
    {
      int j0 = bid * 32 + nt * 16;
      v8f zero = {};
      v8f gi[3], gh[3];
#pragma unroll
      for (int g = 0; g < 3; ++g) { gi[g] = zero; gh[g] = zero; }
      for (int k0 = 0; k0 < kGPAD; k0 += 32) {
        v16bf a = load_a16(gxb, kGPAD, mt * 16, k0, lane);
#pragma unroll
        for (int g = 0; g < 3; ++g) {
          v16bf b = load_b16(wi0, kGPAD, g * 1024 + j0, k0, lane);
          gi[g] = wmma_bf16(a, b, gi[g]);
        }
      }
      const __bf16* h0prev = h0b + (size_t)rp * 64 * 1024;
      for (int k0 = 0; k0 < 1024; k0 += 32) {
        v16bf a = load_a16(h0prev, 1024, mt * 16, k0, lane);
#pragma unroll
        for (int g = 0; g < 3; ++g) {
          v16bf b = load_b16(wh0, 1024, g * 1024 + j0, k0, lane);
          gh[g] = wmma_bf16(a, b, gh[g]);
        }
      }
#pragma unroll
      for (int i = 0; i < 8; ++i) {
        int row = mt * 16 + i + fr, col = j0 + fc;
        float ir = gi[0][i] + bih0[col],        hr = gh[0][i] + bhh0[col];
        float iz = gi[1][i] + bih0[1024 + col], hz = gh[1][i] + bhh0[1024 + col];
        float in_ = gi[2][i] + bih0[2048 + col], hn = gh[2][i] + bhh0[2048 + col];
        float r = sigm_f(ir + hr), z = sigm_f(iz + hz);
        float n = tanhf(in_ + r * hn);
        float hp = h0f[(size_t)rp * 64 * 1024 + (size_t)row * 1024 + col];
        float h = (1.f - z) * n + z * hp;
        h0f[(size_t)wp * 64 * 1024 + (size_t)row * 1024 + col] = h;
        h0b[(size_t)wp * 64 * 1024 + (size_t)row * 1024 + col] = (__bf16)h;
      }
    }
    grid_sync(bar);

    // ---- Phase C: GRU layer 1 (A = h0n bf16, recurrent = h1 prev).
    {
      int j0 = bid * 32 + nt * 16;
      v8f zero = {};
      v8f gi[3], gh[3];
#pragma unroll
      for (int g = 0; g < 3; ++g) { gi[g] = zero; gh[g] = zero; }
      const __bf16* h0cur  = h0b + (size_t)wp * 64 * 1024;
      const __bf16* h1prev = h1b + (size_t)rp * 64 * 1024;
      for (int k0 = 0; k0 < 1024; k0 += 32) {
        v16bf a = load_a16(h0cur, 1024, mt * 16, k0, lane);
#pragma unroll
        for (int g = 0; g < 3; ++g) {
          v16bf b = load_b16(wi1, 1024, g * 1024 + j0, k0, lane);
          gi[g] = wmma_bf16(a, b, gi[g]);
        }
      }
      for (int k0 = 0; k0 < 1024; k0 += 32) {
        v16bf a = load_a16(h1prev, 1024, mt * 16, k0, lane);
#pragma unroll
        for (int g = 0; g < 3; ++g) {
          v16bf b = load_b16(wh1, 1024, g * 1024 + j0, k0, lane);
          gh[g] = wmma_bf16(a, b, gh[g]);
        }
      }
#pragma unroll
      for (int i = 0; i < 8; ++i) {
        int row = mt * 16 + i + fr, col = j0 + fc;
        float ir = gi[0][i] + bih1[col],        hr = gh[0][i] + bhh1[col];
        float iz = gi[1][i] + bih1[1024 + col], hz = gh[1][i] + bhh1[1024 + col];
        float in_ = gi[2][i] + bih1[2048 + col], hn = gh[2][i] + bhh1[2048 + col];
        float r = sigm_f(ir + hr), z = sigm_f(iz + hz);
        float n = tanhf(in_ + r * hn);
        float hp = h1f[(size_t)rp * 64 * 1024 + (size_t)row * 1024 + col];
        float h = (1.f - z) * n + z * hp;
        h1f[(size_t)wp * 64 * 1024 + (size_t)row * 1024 + col] = h;
        h1b[(size_t)wp * 64 * 1024 + (size_t)row * 1024 + col] = (__bf16)h;
      }
    }
    grid_sync(bar);

    // ---- Phase D: layernorm. Block owns rows {2b, 2b+1}: stats + bf16 out.
    {
      const float* hw = h1f + (size_t)wp * 64 * 1024;
      int rl = tid >> 7, cl = tid & 127;
      const float* hrow = hw + (size_t)(2 * bid + rl) * 1024;
      float s = 0.f, q = 0.f;
      for (int c = cl; c < 1024; c += 128) { float v = hrow[c]; s += v; q += v * v; }
      s_sum[tid] = s; s_sq[tid] = q;
      __syncthreads();
      for (int off = 64; off > 0; off >>= 1) {
        if (cl < off) { s_sum[tid] += s_sum[tid + off]; s_sq[tid] += s_sq[tid + off]; }
        __syncthreads();
      }
      for (int i = tid; i < 2048; i += TPB) {
        int r = i >> 10, c = i & 1023;
        float mm = s_sum[r << 7] * (1.f / 1024.f);
        float vv = s_sq[r << 7] * (1.f / 1024.f) - mm * mm;
        float rs = rsqrtf(vv + 1e-5f);
        float v = (hw[(size_t)(2 * bid + r) * 1024 + c] - mm) * rs * ln_g[c] + ln_b[c];
        outb[(size_t)(2 * bid + r) * 1024 + c] = (__bf16)v;
      }
    }
    grid_sync(bar);

    // ---- Phase E: pred = out @ pred_w + b -> mo_hat[:, :, t] and pre_mo.
    if (bid < 8 && nt == 0) {
      int n0 = bid * 16;
      v8f acc = {};
      for (int k0 = 0; k0 < 1024; k0 += 32) {
        v16bf a = load_a16(outb, 1024, mt * 16, k0, lane);
        v16bf b = load_b16(wpred, 1024, n0, k0, lane);
        acc = wmma_bf16(a, b, acc);
      }
#pragma unroll
      for (int i = 0; i < 8; ++i) {
        int row = mt * 16 + i + fr, col = n0 + fc;
        if (col < 126) {
          float p = acc[i] + pred_b[col];
          mo_out[(size_t)row * 126 * kT + (size_t)col * kT + t] = p;
          pre_mo[row * 126 + col] = p;
        }
      }
    }
    grid_sync(bar);
  }
}

// ---------------------------------------------------------------------------
extern "C" void kernel_launch(void* const* d_in, const int* in_sizes, int n_in,
                              void* d_out, int out_size, void* d_ws,
                              size_t ws_size, hipStream_t stream) {
  const float* aud    = (const float*)d_in[0];
  const float* mo     = (const float*)d_in[1];
  const float* lxm    = (const float*)d_in[2];
  const int*   vid    = (const int*)d_in[3];
  const float* eps    = (const float*)d_in[4];
  const float* csi_w0 = (const float*)d_in[5];
  const float* csi_b0 = (const float*)d_in[6];
  const float* csi_w1 = (const float*)d_in[7];
  const float* csi_b1 = (const float*)d_in[8];
  const float* csi_w2 = (const float*)d_in[9];
  const float* csi_b2 = (const float*)d_in[10];
  const float* spk    = (const float*)d_in[11];
  const float* mu_w   = (const float*)d_in[12];
  const float* mu_b   = (const float*)d_in[13];
  const float* lv_w   = (const float*)d_in[14];
  const float* lv_b   = (const float*)d_in[15];
  const float* emb_w  = (const float*)d_in[16];
  const float* emb_b  = (const float*)d_in[17];
  const float* wih0   = (const float*)d_in[18];
  const float* whh0   = (const float*)d_in[19];
  const float* bih0   = (const float*)d_in[20];
  const float* bhh0   = (const float*)d_in[21];
  const float* wih1   = (const float*)d_in[22];
  const float* whh1   = (const float*)d_in[23];
  const float* bih1   = (const float*)d_in[24];
  const float* bhh1   = (const float*)d_in[25];
  const float* ln_g   = (const float*)d_in[26];
  const float* ln_b   = (const float*)d_in[27];
  const float* pred_w = (const float*)d_in[28];
  const float* pred_b = (const float*)d_in[29];

  float* out_mo = (float*)d_out;
  float* out_mu = out_mo + (size_t)64 * 126 * kT;
  float* out_lv = out_mu + (size_t)64 * 32 * kT;

  char* p = (char*)d_ws;
  auto take = [&](size_t bytes) -> char* {
    char* r = p;
    p += (bytes + 255) & ~(size_t)255;
    return r;
  };
  int*    bar   = (int*)take(256);
  __bf16* wc0   = (__bf16*)take((size_t)1024 * 224 * 2);
  __bf16* wc1   = (__bf16*)take((size_t)1024 * 1024 * 2);
  __bf16* wc2   = (__bf16*)take((size_t)2048 * 1024 * 2);
  __bf16* wemb  = (__bf16*)take((size_t)1024 * kXPAD * 2);
  __bf16* wi0   = (__bf16*)take((size_t)3072 * kGPAD * 2);
  __bf16* wh0   = (__bf16*)take((size_t)3072 * 1024 * 2);
  __bf16* wi1   = (__bf16*)take((size_t)3072 * 1024 * 2);
  __bf16* wh1   = (__bf16*)take((size_t)3072 * 1024 * 2);
  __bf16* wpred = (__bf16*)take((size_t)128 * 1024 * 2);
  __bf16* a_x0  = (__bf16*)take((size_t)64 * 224 * 2);
  __bf16* a_h   = (__bf16*)take((size_t)64 * 1024 * 2);
  __bf16* gxb   = (__bf16*)take((size_t)64 * kGPAD * 2);
  __bf16* h0b   = (__bf16*)take((size_t)2 * 64 * 1024 * 2);
  __bf16* h1b   = (__bf16*)take((size_t)2 * 64 * 1024 * 2);
  __bf16* outb  = (__bf16*)take((size_t)64 * 1024 * 2);
  float*  h0f   = (float*)take((size_t)2 * 64 * 1024 * 4);
  float*  h1f   = (float*)take((size_t)2 * 64 * 1024 * 4);
  float*  style = (float*)take((size_t)kT * 64 * 32 * 4);
  float*  premo = (float*)take((size_t)64 * 126 * 4);

  prep_kernel<<<512, TPB, 0, stream>>>(
      mo, lxm, vid, eps, csi_w0, csi_w1, csi_w2, emb_w, wih0, whh0, wih1, whh1,
      pred_w, spk, mu_w, mu_b, lv_w, lv_b, wc0, wc1, wc2, wemb, wi0, wh0, wi1,
      wh1, wpred, a_x0, premo, style, out_mu, out_lv, bar);

  // CSI init-MLP: h = gelu(x0@W0) ; h = gelu(h@W1) ; [h0|h1] = h@W2
  csi_gemm<<<32, TPB, 0, stream>>>(a_x0, 224, 7, wc0, csi_b0, 0, a_h, 1024,
                                   nullptr, nullptr, nullptr, nullptr);
  csi_gemm<<<32, TPB, 0, stream>>>(a_h, 1024, 32, wc1, csi_b1, 0, gxb, 1024,
                                   nullptr, nullptr, nullptr, nullptr);
  csi_gemm<<<64, TPB, 0, stream>>>(gxb, 1024, 32, wc2, csi_b2, 1, nullptr, 0,
                                   h0f, h0b, h1f, h1b);

  scan_kernel<<<NB, TPB, 0, stream>>>(
      aud, lxm, emb_b, bih0, bhh0, bih1, bhh1, ln_g, ln_b, pred_b, wemb, wi0,
      wh0, wi1, wh1, wpred, style, premo, gxb, h0b, h1b, h0f, h1f, outb,
      out_mo, bar);
}
